// DIMKT_54915451847290
// MI455X (gfx1250) — matrix operand order repaired and checked by reading, compiled, not run on hardware
//
#include <hip/hip_runtime.h>
#include <hip/hip_bf16.h>

typedef __attribute__((ext_vector_type(16))) _Float16 v16h;
typedef __attribute__((ext_vector_type(8)))  float    v8f;

union H8 { uint4 u; _Float16 h[8]; };

// ---- WMMA fragment loaders per CDNA5 ISA 7.12.2 (wave32) -------------------
// A (16x32 f16): lane l -> row l%16; VGPR0-3 hold K = 8*(l/16) .. +8,
//                VGPR4-7 hold K = 16 + 8*(l/16) .. +8   (two 16B chunks)
__device__ __forceinline__ v16h load_frag_a(const _Float16* rowp, int lane) {
    int g = (lane >> 4) * 8;
    union { uint4 u[2]; v16h v; } x;
    x.u[0] = *(const uint4*)(rowp + g);
    x.u[1] = *(const uint4*)(rowp + 16 + g);
    return x.v;
}
// B (32x16 f16): lane l -> column l%16; VGPR0-3 hold K = 8*(l/16) .. +8,
//                VGPR4-7 hold K = 16 + 8*(l/16) .. +8   (same order as A)
__device__ __forceinline__ v16h load_frag_b(const _Float16* colp, int lane) {
    int g = (lane >> 4) * 8;
    union { uint4 u[2]; v16h v; } x;
    x.u[0] = *(const uint4*)(colp + g);
    x.u[1] = *(const uint4*)(colp + 16 + g);
    return x.v;
}
__device__ __forceinline__ v8f wmma_f16(v16h a, v16h b, v8f c) {
    return __builtin_amdgcn_wmma_f32_16x16x32_f16(false, a, false, b, (short)0, c, false, false);
}
__device__ __forceinline__ float sigm(float x) { return 1.0f / (1.0f + __expf(-x)); }
__device__ __forceinline__ float tanh_fast(float x) {
    float e = __expf(-2.0f * x);
    return (1.0f - e) / (1.0f + e);
}

#define Bsz 256
#define Ssz 500
#define Esz 128

// ---- Kernel A: fp32 -> fp16 weight conversion ------------------------------
__global__ void cvt_f32_f16(const float* __restrict__ src, _Float16* __restrict__ dst, int n) {
    int i = blockIdx.x * 256 + threadIdx.x;
    if (i < n) dst[i] = (_Float16)src[i];
}

// ---- Kernel B: input GEMM  [S*B,512] @ W1^T -> inpH fp16 [S][B][E] ---------
__global__ __launch_bounds__(256) void embed_gemm1(
    const float* __restrict__ q_emb,            // [B,S,E]
    const int* __restrict__ cI, const int* __restrict__ sdI, const int* __restrict__ qdI, // [B,S]
    const float* __restrict__ c_tab, const float* __restrict__ sd_tab, const float* __restrict__ qd_tab,
    const _Float16* __restrict__ W1h,           // [128,512] fp16
    const float* __restrict__ b1,               // [128]
    _Float16* __restrict__ inpH)                // [S*B,128] fp16, row = t*B + b
{
    __shared__ _Float16 Ah[16 * 528];           // 16 rows x 512 (+16 pad)
    const int tid = threadIdx.x;
    const int rowbase = blockIdx.x * 16;

    // gather A tile: thread -> (row r, 32-half chunk)
    {
        int r  = tid >> 4;
        int k0 = (tid & 15) * 32;
        int row = rowbase + r;
        int t = row >> 8;            // B = 256
        int b = row & 255;
        int seg = k0 >> 7, ko = k0 & 127;
        const float* src; bool zero = false;
        if (seg == 0) {
            src = q_emb + ((long)b * Ssz + t) * Esz + ko;
        } else if (seg == 1) {
            int idx = cI[b * Ssz + t];  zero = (idx == 0); src = c_tab  + (long)idx * Esz + ko;
        } else if (seg == 2) {
            int idx = sdI[b * Ssz + t]; zero = (idx == 0); src = sd_tab + (long)idx * Esz + ko;
        } else {
            int idx = qdI[b * Ssz + t]; zero = (idx == 0); src = qd_tab + (long)idx * Esz + ko;
        }
        _Float16* dst = Ah + r * 528 + k0;
        #pragma unroll
        for (int j = 0; j < 8; j++) {
            float4 v = zero ? make_float4(0.f, 0.f, 0.f, 0.f) : ((const float4*)src)[j];
            dst[j*4+0] = (_Float16)v.x; dst[j*4+1] = (_Float16)v.y;
            dst[j*4+2] = (_Float16)v.z; dst[j*4+3] = (_Float16)v.w;
        }
    }
    __syncthreads();

    const int lane  = tid & 31;
    const int nbase = (tid >> 5) * 16;
    const int ncol  = nbase + (lane & 15);
    const _Float16* arow = Ah  + (lane & 15) * 528;
    const _Float16* bcol = W1h + (long)ncol * 512;
    v8f acc = {};
    #pragma unroll
    for (int kk = 0; kk < 16; kk++) {
        v16h af = load_frag_a(arow + kk * 32, lane);
        v16h bf = load_frag_b(bcol + kk * 32, lane);
        acc = wmma_f16(af, bf, acc);
    }
    float bias = b1[ncol];
    int mh = (lane >> 4) * 8;
    #pragma unroll
    for (int i = 0; i < 8; i++) {
        int row = rowbase + mh + i;
        inpH[(long)row * Esz + ncol] = (_Float16)(acc[i] + bias);
    }
}

// ---- Kernel C: sequential 500-step scan, 16 batch rows per WG --------------
// dynamic LDS layout (bytes):
//  W2L 0..34816  W3L ..69632  W4L ..137216  W5L ..204800
//  kS(f32) ..212992  kH ..217344  qqH ..221696  sdftH ..226048
//  aH ..230400  sdH ..234752  qdH ..239104
#define SCAN_SMEM 239104

__global__ __launch_bounds__(256) void dimkt_scan(
    const _Float16* __restrict__ inpH,
    const int* __restrict__ aI, const int* __restrict__ sdI, const int* __restrict__ qdI,
    const float* __restrict__ a_tab, const float* __restrict__ sd_tab, const float* __restrict__ qd_tab,
    const float* __restrict__ knowledge,
    const _Float16* __restrict__ W2h, const _Float16* __restrict__ W3h,
    const _Float16* __restrict__ W4h, const _Float16* __restrict__ W5h,
    const _Float16* __restrict__ W6h,
    const float* __restrict__ b2, const float* __restrict__ b3, const float* __restrict__ b4,
    const float* __restrict__ b5, const float* __restrict__ b6,
    float* __restrict__ out)
{
    extern __shared__ char smem[];
    _Float16* W2L   = (_Float16*)(smem);
    _Float16* W3L   = (_Float16*)(smem + 34816);
    _Float16* W4L   = (_Float16*)(smem + 69632);
    _Float16* W5L   = (_Float16*)(smem + 137216);
    float*    kS    = (float*)   (smem + 204800);
    _Float16* kH    = (_Float16*)(smem + 212992);
    _Float16* qqH   = (_Float16*)(smem + 217344);
    _Float16* sdftH = (_Float16*)(smem + 221696);
    _Float16* aH    = (_Float16*)(smem + 226048);
    _Float16* sdH   = (_Float16*)(smem + 230400);
    _Float16* qdH   = (_Float16*)(smem + 234752);

    const int tid = threadIdx.x;
    const int rowbase = blockIdx.x * 16;

    // stage W2/W3 (stride 136) and W4/W5 (stride 264) into LDS
    for (int ch = tid; ch < 2048; ch += 256) {
        int n = ch >> 4, kc = (ch & 15) * 8;
        *(uint4*)(W2L + n*136 + kc) = *(const uint4*)(W2h + n*128 + kc);
        *(uint4*)(W3L + n*136 + kc) = *(const uint4*)(W3h + n*128 + kc);
    }
    for (int ch = tid; ch < 4096; ch += 256) {
        int n = ch >> 5, kc = (ch & 31) * 8;
        *(uint4*)(W4L + n*264 + kc) = *(const uint4*)(W4h + n*256 + kc);
        *(uint4*)(W5L + n*264 + kc) = *(const uint4*)(W5h + n*256 + kc);
    }
    // init state k0 = knowledge broadcast
    {
        int m = tid >> 4, kc = (tid & 15) * 8;
        #pragma unroll
        for (int j = 0; j < 8; j++) {
            float v = knowledge[kc + j];
            kS[m*128 + kc + j] = v;
            kH[m*136 + kc + j] = (_Float16)v;
        }
    }

    const int lane  = tid & 31;
    const int nbase = (tid >> 5) * 16;
    const int ncol  = nbase + (lane & 15);
    const int mh    = (lane >> 4) * 8;
    const float bias2 = b2[ncol], bias3 = b3[ncol], bias4 = b4[ncol];
    const float bias5 = b5[ncol], bias6 = b6[ncol];

    // W6 B-fragments are loop-invariant per lane: keep in registers (128 VGPRs)
    v16h w6f[16];
    {
        const _Float16* col = W6h + (long)ncol * 512;
        #pragma unroll
        for (int kk = 0; kk < 16; kk++) w6f[kk] = load_frag_b(col + kk * 32, lane);
    }

    const int em  = tid >> 4;
    const int ekc = (tid & 15) * 8;
    const int eb  = rowbase + em;

    for (int t = 0; t < Ssz; t++) {
        __syncthreads();   // prev-step k writes -> this step's reads
        // ---- stage A: qq = k - inp_t ; gather a/sd/qd embeddings -----------
        {
            int ai = aI[eb * Ssz + t];
            int si = sdI[eb * Ssz + t];
            int qi = qdI[eb * Ssz + t];
            H8 ih; ih.u = *(const uint4*)(inpH + ((long)t * Bsz + eb) * Esz + ekc);
            #pragma unroll
            for (int j = 0; j < 8; j++) {
                float q = kS[em*128 + ekc + j] - (float)ih.h[j];
                qqH[em*136 + ekc + j] = (_Float16)q;
            }
            const float4* ap = (const float4*)(a_tab + (long)ai * Esz + ekc);
            const float4* sp = (const float4*)(sd_tab + (long)si * Esz + ekc);
            const float4* qp = (const float4*)(qd_tab + (long)qi * Esz + ekc);
            #pragma unroll
            for (int j = 0; j < 2; j++) {
                float4 av = ap[j];
                float4 sv = (si == 0) ? make_float4(0.f,0.f,0.f,0.f) : sp[j];
                float4 qv = (qi == 0) ? make_float4(0.f,0.f,0.f,0.f) : qp[j];
                int o = em*136 + ekc + j*4;
                aH[o+0]=(_Float16)av.x; aH[o+1]=(_Float16)av.y; aH[o+2]=(_Float16)av.z; aH[o+3]=(_Float16)av.w;
                sdH[o+0]=(_Float16)sv.x; sdH[o+1]=(_Float16)sv.y; sdH[o+2]=(_Float16)sv.z; sdH[o+3]=(_Float16)sv.w;
                qdH[o+0]=(_Float16)qv.x; qdH[o+1]=(_Float16)qv.y; qdH[o+2]=(_Float16)qv.z; qdH[o+3]=(_Float16)qv.w;
            }
            if (t + 1 < Ssz)
                __builtin_prefetch(inpH + ((long)(t+1) * Bsz + eb) * Esz + ekc, 0, 1);
        }
        __syncthreads();
        // ---- GEMMs W2, W3 over qq (K=128) ----------------------------------
        v8f acc2 = {}, acc3 = {};
        #pragma unroll
        for (int kk = 0; kk < 4; kk++) {
            v16h af  = load_frag_a(qqH + (lane & 15)*136 + kk*32, lane);
            v16h bf2 = load_frag_b(W2L + ncol*136 + kk*32, lane);
            v16h bf3 = load_frag_b(W3L + ncol*136 + kk*32, lane);
            acc2 = wmma_f16(af, bf2, acc2);
            acc3 = wmma_f16(af, bf3, acc3);
        }
        // sdft = sigmoid(.)*tanh(.)
        #pragma unroll
        for (int i = 0; i < 8; i++) {
            float g  = sigm(acc2[i] + bias2);
            float th = tanh_fast(acc3[i] + bias3);
            sdftH[(mh + i)*136 + ncol] = (_Float16)(g * th);
        }
        __syncthreads();
        // ---- GEMMs W4, W5 over xcat=[sdft, a] (K=256) ----------------------
        v8f acc4 = {}, acc5 = {};
        #pragma unroll
        for (int kk = 0; kk < 8; kk++) {
            const _Float16* ab = (kk < 4) ? (sdftH + (lane & 15)*136 + kk*32)
                                          : (aH    + (lane & 15)*136 + (kk - 4)*32);
            v16h af  = load_frag_a(ab, lane);
            v16h bf4 = load_frag_b(W4L + ncol*264 + kk*32, lane);
            v16h bf5 = load_frag_b(W5L + ncol*264 + kk*32, lane);
            acc4 = wmma_f16(af, bf4, acc4);
            acc5 = wmma_f16(af, bf5, acc5);
        }
        // ---- GEMM W6 over ins=[k, a, sd, qd] (K=512), B from registers -----
        v8f acc6 = {};
        #pragma unroll
        for (int kk = 0; kk < 16; kk++) {
            const _Float16* base = (kk < 4) ? kH : (kk < 8) ? aH : (kk < 12) ? sdH : qdH;
            v16h af = load_frag_a(base + (lane & 15)*136 + (kk & 3)*32, lane);
            acc6 = wmma_f16(af, w6f[kk], acc6);
        }
        __syncthreads();   // all kH reads done before k update writes
        // ---- gate update: k = g*k + (1-g)*pkat ------------------------------
        #pragma unroll
        for (int i = 0; i < 8; i++) {
            int m = mh + i;
            float gp   = sigm(acc4[i] + bias4);
            float pkat = gp * tanh_fast(acc5[i] + bias5);
            float g    = sigm(acc6[i] + bias6);
            float kold = kS[m*128 + ncol];
            float knew = g * kold + (1.0f - g) * pkat;
            kS[m*128 + ncol] = knew;
            kH[m*136 + ncol] = (_Float16)knew;
        }
    }
    __syncthreads();
    {
        int m = tid >> 4, kc = (tid & 15) * 8;
        #pragma unroll
        for (int j = 0; j < 8; j++) {
            float v = kS[m*128 + kc + j];
            out[(long)(rowbase + m) * Esz + kc + j] = 1.0f / (1.0f + __expf(-v));
        }
    }
}

// ---- host glue -------------------------------------------------------------
extern "C" void kernel_launch(void* const* d_in, const int* in_sizes, int n_in,
                              void* d_out, int out_size, void* d_ws, size_t ws_size,
                              hipStream_t stream) {
    const int*   cI      = (const int*)  d_in[1];
    const int*   sdI     = (const int*)  d_in[2];
    const int*   qdI     = (const int*)  d_in[3];
    const int*   aI      = (const int*)  d_in[4];
    const float* q_emb   = (const float*)d_in[9];
    const float* c_tab   = (const float*)d_in[10];
    const float* sd_tab  = (const float*)d_in[11];
    const float* qd_tab  = (const float*)d_in[12];
    const float* a_tab   = (const float*)d_in[13];
    const float* knowl   = (const float*)d_in[14];
    const float* W1 = (const float*)d_in[15]; const float* b1 = (const float*)d_in[16];
    const float* W2 = (const float*)d_in[17]; const float* b2 = (const float*)d_in[18];
    const float* W3 = (const float*)d_in[19]; const float* b3 = (const float*)d_in[20];
    const float* W4 = (const float*)d_in[21]; const float* b4 = (const float*)d_in[22];
    const float* W5 = (const float*)d_in[23]; const float* b5 = (const float*)d_in[24];
    const float* W6 = (const float*)d_in[25]; const float* b6 = (const float*)d_in[26];

    _Float16* ws   = (_Float16*)d_ws;
    _Float16* W1h  = ws;                 // 65536
    _Float16* W2h  = ws + 65536;         // 16384
    _Float16* W3h  = ws + 81920;         // 16384
    _Float16* W4h  = ws + 98304;         // 32768
    _Float16* W5h  = ws + 131072;        // 32768
    _Float16* W6h  = ws + 163840;        // 65536
    _Float16* inpH = ws + 229376;        // 500*256*128 fp16, row = t*B+b

    cvt_f32_f16<<<(65536 + 255) / 256, 256, 0, stream>>>(W1, W1h, 65536);
    cvt_f32_f16<<<(16384 + 255) / 256, 256, 0, stream>>>(W2, W2h, 16384);
    cvt_f32_f16<<<(16384 + 255) / 256, 256, 0, stream>>>(W3, W3h, 16384);
    cvt_f32_f16<<<(32768 + 255) / 256, 256, 0, stream>>>(W4, W4h, 32768);
    cvt_f32_f16<<<(32768 + 255) / 256, 256, 0, stream>>>(W5, W5h, 32768);
    cvt_f32_f16<<<(65536 + 255) / 256, 256, 0, stream>>>(W6, W6h, 65536);

    embed_gemm1<<<(Ssz * Bsz) / 16, 256, 0, stream>>>(
        q_emb, cI, sdI, qdI, c_tab, sd_tab, qd_tab, W1h, b1, inpH);

    hipFuncSetAttribute((const void*)dimkt_scan,
                        hipFuncAttributeMaxDynamicSharedMemorySize, SCAN_SMEM);
    dimkt_scan<<<Bsz / 16, 256, SCAN_SMEM, stream>>>(
        inpH, aI, sdI, qdI, a_tab, sd_tab, qd_tab, knowl,
        W2h, W3h, W4h, W5h, W6h, b2, b3, b4, b5, b6, (float*)d_out);
}
